// Word_fusion_block_72018011619645
// MI455X (gfx1250) — matrix operand order repaired and checked by reading, not compiled, not run on hardware
//
#include <hip/hip_runtime.h>
#include <hip/hip_bf16.h>

// ---------------------------------------------------------------------------
// MI455X (gfx1250) implementation.
//   * all GEMMs via v_wmma_f32_16x16x32_bf16 (wave32, 16x16 tiles, f32 accum)
//   * scores -> gumbel -> argmax fused in registers (dots never hit HBM)
//   * LDS tiles staged by the Tensor Data Mover, double-buffered: wave 0
//     (scalar-branch leader) issues tile sp+1, s_wait_tensorcnt on tile sp
//   * one-hot @ v  ==  row gather (forward value of straight-through gumbel)
// ---------------------------------------------------------------------------

typedef unsigned short u16;
typedef unsigned int   u32;
typedef unsigned long long u64;
typedef __attribute__((ext_vector_type(16))) __bf16 v16bf;
typedef __attribute__((ext_vector_type(8)))  float  v8f;
typedef __attribute__((ext_vector_type(4)))  u32    u32x4;
typedef __attribute__((ext_vector_type(8)))  u32    u32x8;
typedef __attribute__((ext_vector_type(4)))  int    i32x4;
typedef __attribute__((ext_vector_type(8)))  int    i32x8;

#define BQ       4
#define NQ       128
#define QDIM     512
#define EMB      512
#define HEADS    8
#define DH       64
#define VOCAB    20000
#define WDIM     300
#define WDIM_PAD 320
#define M1       (BQ * NQ)      // 512 query rows total
#define SCALE    0.125f         // DH^-0.5
#define NCHUNK   8              // vocab chunks for the fused argmax
#define CHUNKN   2560           // 20 super-tiles of 128 cols; 8*2560 >= 20000
#define NSUPER   (CHUNKN / 128) // 20

#if defined(__HIP_DEVICE_COMPILE__) && __has_builtin(__builtin_amdgcn_tensor_load_to_lds)
#define HAVE_TDM 1
#else
#define HAVE_TDM 0
#endif

__device__ __forceinline__ u16 f2bf(float f) {
  u32 u = __float_as_uint(f);
  u32 r = (u + 0x7FFFu + ((u >> 16) & 1u)) >> 16;   // round-to-nearest-even
  return (u16)r;
}

// Gumbel(0,1) noise keyed on the flat 32-bit logit index (threefry stand-in,
// key 42).  u and -ln(u) are guaranteed normal-range floats, so use the raw
// hardware v_log_f32 (log2) without libm's denorm/inf fixup code.
__device__ __forceinline__ float gumbel_noise(u32 id) {
  u32 x = id * 0x9E3779B9u + 0x2A6C1F4Du;  // golden-ratio stir + key-42 salt
  x ^= x >> 16; x *= 0x7FEB352Du;
  x ^= x >> 15; x *= 0x846CA68Bu;
  x ^= x >> 16;
  float u = ((float)(x >> 8) + 0.5f) * (1.0f / 16777216.0f);  // (2^-25, 1)
  const float LN2 = 0.6931471805599453f;
  float t = -LN2 * __builtin_amdgcn_logf(u);   // -ln(u)  in (6e-8, 17.4]
  return -LN2 * __builtin_amdgcn_logf(t);      // -ln(-ln(u))
}

__device__ __forceinline__ v16bf make_frag(u32x4 lo, u32x4 hi) {
  u32x8 r;
  r[0] = lo[0]; r[1] = lo[1]; r[2] = lo[2]; r[3] = lo[3];
  r[4] = hi[0]; r[5] = hi[1]; r[6] = hi[2]; r[7] = hi[3];
  return __builtin_bit_cast(v16bf, r);
}

__device__ __forceinline__ v8f wmma_bf16(v16bf a, v16bf b, v8f c) {
  return __builtin_amdgcn_wmma_f32_16x16x32_bf16(
      /*neg_a=*/false, a, /*neg_b=*/false, b,
      /*c_mod=*/(short)0, c, /*reuse_a=*/false, /*reuse_b=*/false);
}

#if HAVE_TDM
// TDM 2D tile load: bf16 tile (tile_x elems wide, tile_y rows) from a row-
// strided global tensor into LDS at byte offset lds_off.  D# packed per
// CDNA5 ISA ch.8: group0 = {count/type/addr}, group1 = {dims/strides}.
// Rows past tensor_y read as zero (OOB returns zero) -> tail handling free.
__device__ __forceinline__ void tdm_load_tile_bf16(
    u32 lds_off, const void* gptr, u32 tile_x, u32 tile_y,
    u32 tensor_x, u32 tensor_y, u32 row_stride_elems) {
  u64 ga = (u64)(uintptr_t)gptr;
  u32x4 g0;
  g0[0] = 1u;                                           // count=1, user desc
  g0[1] = lds_off;                                      // lds_addr [63:32]
  g0[2] = (u32)(ga & 0xFFFFFFFFu);                      // global_addr[31:0]
  g0[3] = (u32)((ga >> 32) & 0x01FFFFFFu) | (2u << 30); // ga[56:32], type=2
  i32x8 g1;
  u32 w0 = (1u << 16);                                  // data_size=1 -> 2B
  u32 w1 = (tensor_x & 0xFFFFu) << 16;                  // tensor_dim0 lo16
  u32 w2 = (tensor_x >> 16) | ((tensor_y & 0xFFFFu) << 16);  // dim0 hi / dim1 lo
  u32 w3 = (tensor_y >> 16) | ((tile_x & 0xFFFFu) << 16);    // dim1 hi / tile0
  u32 w4 = (tile_y & 0xFFFFu);                          // tile_dim1 (tile2=0)
  u32 w5 = row_stride_elems;                            // dim0_stride [31:0]
  g1[0] = (int)w0; g1[1] = (int)w1; g1[2] = (int)w2; g1[3] = (int)w3;
  g1[4] = (int)w4; g1[5] = (int)w5; g1[6] = 0; g1[7] = 0;
  i32x4 z4; z4[0] = 0; z4[1] = 0; z4[2] = 0; z4[3] = 0;
#if __clang_major__ >= 23
  i32x8 z8;
#pragma unroll
  for (int i = 0; i < 8; ++i) z8[i] = 0;
  __builtin_amdgcn_tensor_load_to_lds(g0, g1, z4, z4, z8, 0);
#else
  __builtin_amdgcn_tensor_load_to_lds(g0, g1, z4, z4, 0);
#endif
}
#endif

// ---------------------------------------------------------------------------
// Prep: fp32 -> bf16 converts (with K padding / transpose for B operands)
// ---------------------------------------------------------------------------
__global__ void convert_pad_rows_kernel(const float* __restrict__ in,
                                        u16* __restrict__ out,
                                        int cols, int colsPad, int total) {
  for (int o = blockIdx.x * blockDim.x + threadIdx.x; o < total;
       o += gridDim.x * blockDim.x) {
    int r = o / colsPad, c = o - r * colsPad;
    out[o] = (c < cols) ? f2bf(in[(size_t)r * cols + c]) : (u16)0;
  }
}

// W[K][N] (in,out layout; y = x@W) -> WT[N][Kpad] bf16, zero-padded in K.
__global__ void convert_transpose_kernel(const float* __restrict__ W,
                                         u16* __restrict__ WT,
                                         int K, int N, int Kpad) {
  int total = N * Kpad;
  for (int o = blockIdx.x * blockDim.x + threadIdx.x; o < total;
       o += gridDim.x * blockDim.x) {
    int n = o / Kpad, k = o - n * Kpad;
    WT[o] = (k < K) ? f2bf(W[(size_t)k * N + n]) : (u16)0;
  }
}

// ---------------------------------------------------------------------------
// LayerNorm over QDIM=512; one block per row, output bf16.
// ---------------------------------------------------------------------------
__global__ __launch_bounds__(128) void ln_kernel(const float* __restrict__ x,
                                                 const float* __restrict__ gamma,
                                                 const float* __restrict__ beta,
                                                 u16* __restrict__ out) {
  __shared__ float red[8];
  const int row = blockIdx.x;
  const int t = threadIdx.x;
  const int wid = t >> 5, lane = t & 31;
  const float* xr = x + (size_t)row * QDIM;

  float v[4];
  float s = 0.f;
#pragma unroll
  for (int i = 0; i < 4; ++i) { v[i] = xr[t + i * 128]; s += v[i]; }
#pragma unroll
  for (int off = 16; off >= 1; off >>= 1) s += __shfl_xor(s, off, 32);
  if (lane == 0) red[wid] = s;
  __syncthreads();
  float mean = (red[0] + red[1] + red[2] + red[3]) * (1.f / 512.f);

  float ss = 0.f;
#pragma unroll
  for (int i = 0; i < 4; ++i) { float d = v[i] - mean; ss += d * d; }
#pragma unroll
  for (int off = 16; off >= 1; off >>= 1) ss += __shfl_xor(ss, off, 32);
  if (lane == 0) red[4 + wid] = ss;
  __syncthreads();
  float var = (red[4] + red[5] + red[6] + red[7]) * (1.f / 512.f);
  float rs = rsqrtf(var + 1e-5f);

#pragma unroll
  for (int i = 0; i < 4; ++i) {
    int col = t + i * 128;
    out[(size_t)row * QDIM + col] = f2bf((v[i] - mean) * rs * gamma[col] + beta[col]);
  }
}

// ---------------------------------------------------------------------------
// Generic WMMA-bf16 GEMM: C[M,N] = A[M,K](bf16,row) x BT[N,K](bf16,N-major).
// Block tile 128x128, K-step 32, 8 waves (2x4), wave tile 64x32 (4x2 frags).
// A/B k-tiles (8KB each) staged by TDM, double-buffered (LDS: A0 B0 A1 B1).
// EPI: 0 -> bf16 store, 1 -> f32 store, 2 -> f32 + bias[n] + residual[m][n].
// ---------------------------------------------------------------------------
template <int EPI>
__global__ __launch_bounds__(256) void gemm_bf16_kernel(
    const u16* __restrict__ A, int lda,
    const u16* __restrict__ BT, int ldb,
    void* __restrict__ Cout, int ldc,
    int M, int N, int K,
    const float* __restrict__ bias,
    const float* __restrict__ res, int ldr) {
  extern __shared__ u16 SM[];  // [A0:4096][B0:4096][A1:4096][B1:4096] u16
  const int tid = threadIdx.x;
  const int lane = tid & 31, wid = tid >> 5;
  const int half = lane >> 4, l16 = lane & 15;
  const int wm = wid >> 2, wn = wid & 3;
  const int mBase = blockIdx.y * 128;
  const int nBase = blockIdx.x * 128;
  const int nsteps = K / 32;

  v8f acc[4][2];
#pragma unroll
  for (int f = 0; f < 4; ++f)
#pragma unroll
    for (int g = 0; g < 2; ++g)
#pragma unroll
      for (int i = 0; i < 8; ++i) acc[f][g][i] = 0.f;

#if HAVE_TDM
  const bool leader = (__builtin_amdgcn_readfirstlane(wid) == 0);
  if (leader) {
    tdm_load_tile_bf16(0, A + (size_t)mBase * lda, 32, 128, 32,
                       (u32)(M - mBase), (u32)lda);
    tdm_load_tile_bf16(8192, BT + (size_t)nBase * ldb, 32, 128, 32,
                       (u32)(N - nBase), (u32)ldb);
  }
#endif

  for (int it = 0; it < nsteps; ++it) {
    const int kb = it * 32;
    const int cur = it & 1;
    const u16* Abuf;
    const u16* Bbuf;
#if HAVE_TDM
    Abuf = SM + cur * 8192;
    Bbuf = SM + 4096 + cur * 8192;
    if (leader) {
      if (it + 1 < nsteps) {
        const u32 nxt = (u32)(1 - cur);
        tdm_load_tile_bf16(nxt * 16384u,
                           A + (size_t)mBase * lda + kb + 32, 32, 128, 32,
                           (u32)(M - mBase), (u32)lda);
        tdm_load_tile_bf16(8192u + nxt * 16384u,
                           BT + (size_t)nBase * ldb + kb + 32, 32, 128, 32,
                           (u32)(N - nBase), (u32)ldb);
        __builtin_amdgcn_s_wait_tensorcnt(2);  // current A/B tiles complete
      } else {
        __builtin_amdgcn_s_wait_tensorcnt(0);
      }
    }
    __syncthreads();  // current tiles visible to all waves
#else
    Abuf = SM;
    Bbuf = SM + 4096;
    __syncthreads();  // previous tiles fully consumed
#pragma unroll
    for (int jt = 0; jt < 2; ++jt) {
      int c = tid + jt * 256;
      int row = c >> 2, q = c & 3;
      int gm = mBase + row;
      u32x4 va = {0, 0, 0, 0};
      if (gm < M) {
        va = *(const u32x4*)(A + (size_t)gm * lda + kb + q * 8);
        if (kb + 32 < K)
          __builtin_prefetch(A + (size_t)gm * lda + kb + 32 + q * 8, 0, 1);
      }
      *(u32x4*)(SM + row * 32 + q * 8) = va;
      int gn = nBase + row;
      u32x4 vb = {0, 0, 0, 0};
      if (gn < N) {
        vb = *(const u32x4*)(BT + (size_t)gn * ldb + kb + q * 8);
        if (kb + 32 < K)
          __builtin_prefetch(BT + (size_t)gn * ldb + kb + 32 + q * 8, 0, 1);
      }
      *(u32x4*)(SM + 4096 + row * 32 + q * 8) = vb;
    }
    __syncthreads();
#endif

    v16bf afrag[4], bfrag[2];
#pragma unroll
    for (int f = 0; f < 4; ++f) {
      // 16-bit A 16x32 layout: lane half picks K runs {h*8..}, {16+h*8..}
      const u16* p = Abuf + (wm * 64 + f * 16 + l16) * 32;
      afrag[f] = make_frag(*(const u32x4*)(p + half * 8),
                           *(const u32x4*)(p + 16 + half * 8));
    }
#pragma unroll
    for (int g = 0; g < 2; ++g) {
      // 16-bit B 32x16 layout: lane half picks K runs {h*8..}, {16+h*8..}
      const u16* p = Bbuf + (wn * 32 + g * 16 + l16) * 32;
      bfrag[g] = make_frag(*(const u32x4*)(p + half * 8),
                           *(const u32x4*)(p + 16 + half * 8));
    }
#pragma unroll
    for (int f = 0; f < 4; ++f)
#pragma unroll
      for (int g = 0; g < 2; ++g)
        acc[f][g] = wmma_bf16(afrag[f], bfrag[g], acc[f][g]);
    __syncthreads();  // tiles consumed; safe to overwrite this buffer next
  }

  // epilogue; C layout: VGPR i -> M = i + half*8, N = l16
#pragma unroll
  for (int f = 0; f < 4; ++f)
#pragma unroll
    for (int g = 0; g < 2; ++g)
#pragma unroll
      for (int i = 0; i < 8; ++i) {
        int m = mBase + wm * 64 + f * 16 + i + half * 8;
        int n = nBase + wn * 32 + g * 16 + l16;
        if (m >= M) continue;
        float val = acc[f][g][i];
        if constexpr (EPI == 0) {
          ((u16*)Cout)[(size_t)m * ldc + n] = f2bf(val);
        } else if constexpr (EPI == 1) {
          ((float*)Cout)[(size_t)m * ldc + n] = val;
        } else {
          ((float*)Cout)[(size_t)m * ldc + n] =
              val + bias[n] + res[(size_t)m * ldr + n];
        }
      }
}

// ---------------------------------------------------------------------------
// Fused scores + gumbel + running argmax.
// Grid: 256 blocks = b(4) x h(8) x vocab-chunk(8). Block = 8 waves; wave w
// owns query rows [16w, 16w+16). Q fragments live in registers for the whole
// kernel; k tiles (128 vocab rows x 64 dh bf16 = 16KB) double-buffered in
// LDS via TDM. dots never leave registers.
// ---------------------------------------------------------------------------
__global__ __launch_bounds__(256) void scores_argmax_kernel(
    const u16* __restrict__ qb,   // [M1][EMB] bf16
    const u16* __restrict__ kb,   // [VOCAB][EMB] bf16
    float* __restrict__ pvals,    // [BQ*HEADS*NQ][NCHUNK]
    int* __restrict__ pidx) {
  extern __shared__ u16 SM[];     // two 128*DH bf16 buffers (16KB each)
  const int tid = threadIdx.x;
  const int lane = tid & 31, wid = tid >> 5;
  const int half = lane >> 4, l16 = lane & 15;
  const int chunk = blockIdx.x & 7;
  const int h = (blockIdx.x >> 3) & 7;
  const int b = blockIdx.x >> 6;

  v16bf afrag[2];
  {
    const u16* qbase = qb + (size_t)(b * NQ + wid * 16 + l16) * EMB + h * DH;
#pragma unroll
    for (int ks = 0; ks < 2; ++ks)
      afrag[ks] = make_frag(*(const u32x4*)(qbase + ks * 32 + half * 8),
                            *(const u32x4*)(qbase + ks * 32 + 16 + half * 8));
  }

  float bestv[8];
  int besti[8];
#pragma unroll
  for (int i = 0; i < 8; ++i) { bestv[i] = -3.4e38f; besti[i] = 0; }

  // flat logit id base for this (b,h, wave row block); fits in 32 bits
  const u32 idBase = (u32)(((b * HEADS + h) * NQ + wid * 16) * VOCAB) +
                     (u32)(half * 8) * (u32)VOCAB;

  const int chunkBase = chunk * CHUNKN;
#if HAVE_TDM
  const bool leader = (__builtin_amdgcn_readfirstlane(wid) == 0);
  if (leader) {
    int rows = VOCAB - chunkBase;
    tdm_load_tile_bf16(0, kb + (size_t)chunkBase * EMB + h * DH, DH, 128, DH,
                       (u32)(rows > 0 ? rows : 0), EMB);
  }
#endif

  for (int sp = 0; sp < NSUPER; ++sp) {
    const int n0 = chunkBase + sp * 128;
    const int cur = sp & 1;
    const u16* Kbuf;
#if HAVE_TDM
    Kbuf = SM + cur * 8192;
    if (leader) {
      if (sp + 1 < NSUPER) {
        int n1 = n0 + 128;
        int rows1 = VOCAB - n1;
        tdm_load_tile_bf16((u32)(1 - cur) * 16384u,
                           kb + (size_t)n1 * EMB + h * DH, DH, 128, DH,
                           (u32)(rows1 > 0 ? rows1 : 0), EMB);
        __builtin_amdgcn_s_wait_tensorcnt(1);  // current tile complete
      } else {
        __builtin_amdgcn_s_wait_tensorcnt(0);
      }
    }
    __syncthreads();  // current tile visible to all waves
#else
    Kbuf = SM;
    __syncthreads();  // everyone done reading the previous tile
#pragma unroll
    for (int jt = 0; jt < 4; ++jt) {
      int c = tid + jt * 256;
      int row = c >> 3, q8 = c & 7;
      int gn = n0 + row;
      u32x4 v = {0, 0, 0, 0};
      if (gn < VOCAB) {
        v = *(const u32x4*)(kb + (size_t)gn * EMB + h * DH + q8 * 8);
        if (n0 + 128 < VOCAB)
          __builtin_prefetch(kb + (size_t)(gn + 128) * EMB + h * DH + q8 * 8, 0, 1);
      }
      *(u32x4*)(SM + row * DH + q8 * 8) = v;
    }
    __syncthreads();
#endif

#pragma unroll
    for (int nt = 0; nt < 8; ++nt) {
      v8f acc;
#pragma unroll
      for (int i = 0; i < 8; ++i) acc[i] = 0.f;
#pragma unroll
      for (int ks = 0; ks < 2; ++ks) {
        const u16* p = Kbuf + (nt * 16 + l16) * DH + ks * 32;
        acc = wmma_bf16(afrag[ks],
                        make_frag(*(const u32x4*)(p + half * 8), *(const u32x4*)(p + 16 + half * 8)),
                        acc);
      }
      int n = n0 + nt * 16 + l16;
      if (n < VOCAB) {
        u32 idn = idBase + (u32)n;
#pragma unroll
        for (int i = 0; i < 8; ++i) {
          float val = acc[i] * SCALE + gumbel_noise(idn + (u32)i * (u32)VOCAB);
          if (val > bestv[i]) { bestv[i] = val; besti[i] = n; }
        }
      }
    }
#if HAVE_TDM
    __syncthreads();  // tile consumed before TDM overwrites it next round
#endif
  }

  // reduce (max, lowest-index tie-break) across the 16 lanes of each half
#pragma unroll
  for (int i = 0; i < 8; ++i) {
#pragma unroll
    for (int off = 1; off < 16; off <<= 1) {
      float ov = __shfl_xor(bestv[i], off, 32);
      int oi = __shfl_xor(besti[i], off, 32);
      if (ov > bestv[i] || (ov == bestv[i] && oi < besti[i])) {
        bestv[i] = ov; besti[i] = oi;
      }
    }
  }
  if (l16 == 0) {
#pragma unroll
    for (int i = 0; i < 8; ++i) {
      int m = wid * 16 + i + half * 8;
      int p = ((b * HEADS + h) * NQ + m) * NCHUNK + chunk;
      pvals[p] = bestv[i];
      pidx[p] = besti[i];
    }
  }
}

__global__ void argmax_reduce_kernel(const float* __restrict__ pvals,
                                     const int* __restrict__ pidx,
                                     int* __restrict__ idxout) {
  int t = blockIdx.x * blockDim.x + threadIdx.x;
  if (t >= BQ * HEADS * NQ) return;
  float bv = -3.4e38f;
  int bi = 0;
#pragma unroll
  for (int c = 0; c < NCHUNK; ++c) {
    float v = pvals[t * NCHUNK + c];
    if (v > bv) { bv = v; bi = pidx[t * NCHUNK + c]; }  // ascending c keeps lowest idx
  }
  idxout[t] = bi;
}

// one-hot @ v  ->  gather v rows into the out-proj A operand
__global__ __launch_bounds__(64) void gather_kernel(const u16* __restrict__ vb,
                                                    const int* __restrict__ idxb,
                                                    u16* __restrict__ gath) {
  int rq = blockIdx.x;             // b*NQ + q
  int b = rq >> 7, q = rq & 127;
  int t = threadIdx.x;
  int h = t >> 3, c = t & 7;
  int vrow = idxb[(b * HEADS + h) * NQ + q];
  *(u32x4*)(gath + (size_t)rq * EMB + h * DH + c * 8) =
      *(const u32x4*)(vb + (size_t)vrow * EMB + h * DH + c * 8);
}

// ---------------------------------------------------------------------------
extern "C" void kernel_launch(void* const* d_in, const int* in_sizes, int n_in,
                              void* d_out, int out_size, void* d_ws,
                              size_t ws_size, hipStream_t stream) {
  (void)in_sizes; (void)n_in; (void)out_size; (void)ws_size;
  const float* queries = (const float*)d_in[0];  // [4,128,512]
  const float* glove   = (const float*)d_in[1];  // [20000,300]
  const float* Wq      = (const float*)d_in[2];  // [512,512]
  const float* Wk      = (const float*)d_in[3];  // [300,512]
  const float* Wv      = (const float*)d_in[4];  // [300,512]
  const float* Wout    = (const float*)d_in[5];  // [512,512]
  const float* b_out   = (const float*)d_in[6];  // [512]
  const float* ln_g    = (const float*)d_in[7];  // [512]
  const float* ln_b    = (const float*)d_in[8];  // [512]
  float* out = (float*)d_out;                    // [4,128,512] f32

  char* ws = (char*)d_ws;
  size_t off = 0;
  auto carve = [&](size_t bytes) -> char* {
    off = (off + 255) & ~(size_t)255;
    char* p = ws + off;
    off += bytes;
    return p;
  };
  u16* gloveb = (u16*)carve((size_t)VOCAB * WDIM_PAD * 2);
  u16* WqT    = (u16*)carve((size_t)QDIM * QDIM * 2);
  u16* WkT    = (u16*)carve((size_t)EMB * WDIM_PAD * 2);
  u16* WvT    = (u16*)carve((size_t)EMB * WDIM_PAD * 2);
  u16* WoT    = (u16*)carve((size_t)QDIM * EMB * 2);
  u16* xnb    = (u16*)carve((size_t)M1 * QDIM * 2);
  u16* qbf    = (u16*)carve((size_t)M1 * EMB * 2);
  u16* kbf    = (u16*)carve((size_t)VOCAB * EMB * 2);
  u16* vbf    = (u16*)carve((size_t)VOCAB * EMB * 2);
  u16* gath   = (u16*)carve((size_t)M1 * EMB * 2);
  float* pvals = (float*)carve((size_t)BQ * HEADS * NQ * NCHUNK * 4);
  int*   pidx  = (int*)carve((size_t)BQ * HEADS * NQ * NCHUNK * 4);
  int*   idxb  = (int*)carve((size_t)BQ * HEADS * NQ * 4);

  const size_t GEMM_LDS   = 4 * 4096 * sizeof(u16);  // A0 B0 A1 B1 = 32KB
  const size_t SCORES_LDS = 2 * 8192 * sizeof(u16);  // K0 K1       = 32KB

  // --- prep converts ---
  convert_pad_rows_kernel<<<4096, 256, 0, stream>>>(glove, gloveb, WDIM,
                                                    WDIM_PAD, VOCAB * WDIM_PAD);
  convert_transpose_kernel<<<1024, 256, 0, stream>>>(Wq, WqT, QDIM, EMB, QDIM);
  convert_transpose_kernel<<<640, 256, 0, stream>>>(Wk, WkT, WDIM, EMB, WDIM_PAD);
  convert_transpose_kernel<<<640, 256, 0, stream>>>(Wv, WvT, WDIM, EMB, WDIM_PAD);
  convert_transpose_kernel<<<1024, 256, 0, stream>>>(Wout, WoT, EMB, QDIM, EMB);

  // --- LayerNorm ---
  ln_kernel<<<M1, 128, 0, stream>>>(queries, ln_g, ln_b, xnb);

  // --- projections (WMMA bf16, TDM-staged tiles) ---
  gemm_bf16_kernel<0><<<dim3(EMB / 128, M1 / 128), 256, GEMM_LDS, stream>>>(
      xnb, QDIM, WqT, QDIM, qbf, EMB, M1, EMB, QDIM, nullptr, nullptr, 0);
  gemm_bf16_kernel<0>
      <<<dim3(EMB / 128, (VOCAB + 127) / 128), 256, GEMM_LDS, stream>>>(
          gloveb, WDIM_PAD, WkT, WDIM_PAD, kbf, EMB, VOCAB, EMB, WDIM_PAD,
          nullptr, nullptr, 0);
  gemm_bf16_kernel<0>
      <<<dim3(EMB / 128, (VOCAB + 127) / 128), 256, GEMM_LDS, stream>>>(
          gloveb, WDIM_PAD, WvT, WDIM_PAD, vbf, EMB, VOCAB, EMB, WDIM_PAD,
          nullptr, nullptr, 0);

  // --- fused scores + gumbel + argmax (TDM double-buffered k tiles) ---
  scores_argmax_kernel<<<BQ * HEADS * NCHUNK, 256, SCORES_LDS, stream>>>(
      qbf, kbf, pvals, pidx);
  argmax_reduce_kernel<<<(BQ * HEADS * NQ + 255) / 256, 256, 0, stream>>>(
      pvals, pidx, idxb);

  // --- one-hot attention == gather, then out-proj + bias + residual ---
  gather_kernel<<<M1, 64, 0, stream>>>(vbf, idxb, gath);
  gemm_bf16_kernel<2><<<dim3(QDIM / 128, M1 / 128), 256, GEMM_LDS, stream>>>(
      gath, EMB, WoT, EMB, out, QDIM, M1, QDIM, EMB, b_out, queries, QDIM);
}
